// MultiHeadAttention_61357902791248
// MI455X (gfx1250) — compile-verified, matrix operand order checked by reading, not run on hardware
//
#include <hip/hip_runtime.h>

// ---------------------------------------------------------------------------
// MI455X (gfx1250, wave32) fused MHA: bf16 WMMA GEMMs + flash attention.
// Async global->LDS (ASYNCcnt) staging, double-buffered GEMM mainloop.
// ---------------------------------------------------------------------------

typedef __attribute__((ext_vector_type(16))) __bf16 v16bf;
typedef __attribute__((ext_vector_type(8)))  __bf16 v8bf;
typedef __attribute__((ext_vector_type(8)))  float  v8f;

static constexpr int Bc   = 2;
static constexpr int Sc   = 2048;
static constexpr int HIDc = 2048;
static constexpr int Hc   = 16;
static constexpr int Dc   = 128;
static constexpr float QK_SCALE = 0.08838834764831845f; // 1/sqrt(128)

__device__ __forceinline__ __bf16 f2bf(float f) {
    unsigned u = __builtin_bit_cast(unsigned, f);
    u += 0x7FFFu + ((u >> 16) & 1u);           // round-to-nearest-even
    unsigned short h = (unsigned short)(u >> 16);
    return __builtin_bit_cast(__bf16, h);
}
__device__ __forceinline__ float bf2f(__bf16 b) {
    unsigned short h = __builtin_bit_cast(unsigned short, b);
    unsigned u = ((unsigned)h) << 16;
    return __builtin_bit_cast(float, u);
}

__device__ __forceinline__ v8f wmma_bf16(v16bf a, v16bf b, v8f c) {
    return __builtin_amdgcn_wmma_f32_16x16x32_bf16(false, a, false, b, (short)0, c, false, false);
}

// Generic pointer to LDS byte offset (LDS aperture lives in addr[63:32];
// addr[31:0] is the LDS offset per the CDNA5 aperture rules).
__device__ __forceinline__ unsigned lds_off(const void* p) {
    return (unsigned)(unsigned long long)p;
}

// CDNA5 async DMA: 16 B global -> LDS, tracked by ASYNCcnt.
// VGLOBAL encoding: VDST = LDS byte address, VADDR = 64-bit global address.
__device__ __forceinline__ void async_g2l_b128(unsigned lds_addr, const void* gptr) {
    asm volatile("global_load_async_to_lds_b128 %0, %1, off"
                 :: "v"(lds_addr), "v"((unsigned long long)gptr)
                 : "memory");
}
__device__ __forceinline__ void wait_async0() {
    asm volatile("s_wait_asynccnt 0" ::: "memory");
}

// Load a 16(rows) x 32(k) WMMA A/B fragment from a row-major bf16 buffer.
// ISA 16-bit A layout: lanes 0-15 hold row M=lane, k = {0..7,16..23};
// lanes 16-31 hold row M=lane-16, k = {8..15,24..31}.  Two b128 loads/lane.
__device__ __forceinline__ v16bf frag_ld(const __bf16* base, int row0, int col0,
                                         int ld, int lane) {
    const __bf16* p = base + (size_t)(row0 + (lane & 15)) * ld
                           + col0 + ((lane >> 4) << 3);
    v8bf x0 = *(const v8bf*)(p);
    v8bf x1 = *(const v8bf*)(p + 16);
    return __builtin_shufflevector(x0, x1, 0,1,2,3,4,5,6,7,8,9,10,11,12,13,14,15);
}

// ---------------------------------------------------------------------------
// fp32 -> bf16 cast (vectorized)
// ---------------------------------------------------------------------------
__global__ void cast_to_bf16(const float* __restrict__ in, __bf16* __restrict__ out, int n) {
    int i = (blockIdx.x * blockDim.x + threadIdx.x) * 4;
    if (i + 3 < n) {
        float4 v = *(const float4*)(in + i);
        out[i + 0] = f2bf(v.x);
        out[i + 1] = f2bf(v.y);
        out[i + 2] = f2bf(v.z);
        out[i + 3] = f2bf(v.w);
    }
}

// ---------------------------------------------------------------------------
// Repack f32 row-major weight W[K][N] into bf16 WMMA-B-fragment layout:
// frag (nt,kt): 32 lanes x 16 contiguous bf16; lane l<16 -> col n0+l,
// k = k0+{0..7,16..23}; lane l>=16 -> col n0+l-16, k = k0+{8..15,24..31}.
// Fragment index = nt*(K/32)+kt  (kt-minor => K loop streams contiguously).
// ---------------------------------------------------------------------------
__global__ void pack_w(const float* __restrict__ W, __bf16* __restrict__ Bp, int K, int N) {
    int t    = blockIdx.x * blockDim.x + threadIdx.x;
    int lane = t & 31;
    int frag = t >> 5;
    int ktiles = K >> 5;
    int kt = frag % ktiles;
    int nt = frag / ktiles;
    if (nt >= (N >> 4)) return;
    int n     = (nt << 4) + (lane & 15);
    int kbase = (kt << 5) + ((lane >> 4) << 3);
    __bf16* dst = Bp + ((size_t)frag << 9) + (size_t)lane * 16;
#pragma unroll
    for (int j = 0; j < 8; ++j) {
        dst[j]     = f2bf(W[(size_t)(kbase + j)      * N + n]);
        dst[8 + j] = f2bf(W[(size_t)(kbase + 16 + j) * N + n]);
    }
}

// ---------------------------------------------------------------------------
// Tiled bf16 WMMA GEMM: C[M,N] = A[M,K] @ B[K,N]
//   256 threads = 8 waves; wave grid 2(M)x4(N); 32x64 per wave; WG 64x256.
//   Double-buffered LDS fed by global_load_async_to_lds_b128 (ASYNCcnt).
//   mode 0: bf16 row-major (ldc=N)                     -> Q, K projections
//   mode 1: bf16 transposed per head Vt[b][h][d][s]    -> V projection
//   mode 2: f32 row-major                              -> final output
// ---------------------------------------------------------------------------
__device__ __forceinline__ void store_tile(v8f acc, int row0, int col0,
                                           float* Cf, __bf16* Cb, int N,
                                           int mode, int lane) {
    int row = row0 + ((lane >> 4) << 3);
    int col = col0 + (lane & 15);
#pragma unroll
    for (int r = 0; r < 8; ++r) {
        float v = acc[r];
        int m = row + r;
        if (mode == 0) {
            Cb[(size_t)m * N + col] = f2bf(v);
        } else if (mode == 1) {
            int bb = m >> 11, s = m & 2047;
            int hh = col >> 7, d = col & 127;
            Cb[(((size_t)(bb * Hc + hh)) * Dc + d) * Sc + s] = f2bf(v);
        } else {
            Cf[(size_t)m * N + col] = v;
        }
    }
}

__global__ __launch_bounds__(256) void gemm_bf16_wmma(
    const __bf16* __restrict__ A, const __bf16* __restrict__ Bp,
    float* Cf, __bf16* Cb, int M, int N, int K, int mode) {
    __shared__ __bf16 sA[2][64 * 32];    // 2 x 4 KB
    __shared__ __bf16 sB[2][16 * 512];   // 2 x 16 KB (packed frags, 256 cols)

    const int tid  = threadIdx.x;
    const int lane = tid & 31;
    const int wv   = tid >> 5;
    const int wm   = wv >> 2;      // 0..1
    const int wn   = wv & 3;       // 0..3
    const int m0   = blockIdx.y * 64;
    const int n0   = blockIdx.x * 256;
    const int ktiles = K >> 5;

    v8f acc0[4] = {}, acc1[4] = {};

    // async-transfer coordinates (per thread: 16 B of A, 64 B of B)
    const int ar = tid >> 2;                 // 0..63  A row
    const int ac = (tid & 3) * 8;            // A col chunk (8 elts = 16 B)
    const int bf = tid >> 4;                 // 0..15  local n-tile
    const int be = (tid & 15) * 32;          // elt offset inside 512-elt frag

    auto issue = [&](int kt, int buf) {
        unsigned aoff = lds_off(&sA[buf][ar * 32 + ac]);
        async_g2l_b128(aoff, &A[(size_t)(m0 + ar) * K + kt * 32 + ac]);
        unsigned boff = lds_off(&sB[buf][bf * 512 + be]);
        const __bf16* gB = Bp + ((size_t)(blockIdx.x * 16 + bf) * ktiles + kt) * 512 + be;
        async_g2l_b128(boff,      gB);
        async_g2l_b128(boff + 16, gB + 8);
        async_g2l_b128(boff + 32, gB + 16);
        async_g2l_b128(boff + 48, gB + 24);
        if (kt + 1 < ktiles)
            __builtin_prefetch(gB + 512, 0, 1);       // global_prefetch_b8
    };

    issue(0, 0);
    for (int kt = 0; kt < ktiles; ++kt) {
        wait_async0();             // own transfers for buf kt&1 complete
        __syncthreads();           // everyone's transfers complete; prev buf free
        if (kt + 1 < ktiles) issue(kt + 1, (kt + 1) & 1);

        const __bf16* cA = sA[kt & 1];
        const __bf16* cB = sB[kt & 1];
        v16bf a0 = frag_ld(cA, wm * 32,      0, 32, lane);
        v16bf a1 = frag_ld(cA, wm * 32 + 16, 0, 32, lane);
#pragma unroll
        for (int j = 0; j < 4; ++j) {
            v16bf b = *(const v16bf*)&cB[(wn * 4 + j) * 512 + lane * 16];
            acc0[j] = wmma_bf16(a0, b, acc0[j]);
            acc1[j] = wmma_bf16(a1, b, acc1[j]);
        }
    }
#pragma unroll
    for (int j = 0; j < 4; ++j) {
        store_tile(acc0[j], m0 + wm * 32,      n0 + wn * 64 + j * 16, Cf, Cb, N, mode, lane);
        store_tile(acc1[j], m0 + wm * 32 + 16, n0 + wn * 64 + j * 16, Cf, Cb, N, mode, lane);
    }
}

// ---------------------------------------------------------------------------
// RoPE on Q and K (bf16 in-place; each thread owns a (d, d+64) pair -> safe).
// ---------------------------------------------------------------------------
__global__ void rope_qk(__bf16* __restrict__ Q, __bf16* __restrict__ Kk,
                        const float* __restrict__ fc, const float* __restrict__ fs) {
    unsigned t = blockIdx.x * blockDim.x + threadIdx.x;   // 2^22 threads
    int d  = t & 63;
    int hh = (t >> 6) & 15;
    int s  = (t >> 10) & 2047;
    int bb = (t >> 21) & 1;
    size_t base = ((size_t)(bb * Sc + s)) * HIDc + hh * Dc;
    float c1 = fc[s * Dc + d],      s1 = fs[s * Dc + d];
    float c2 = fc[s * Dc + 64 + d], s2 = fs[s * Dc + 64 + d];

    float q1 = bf2f(Q[base + d]), q2 = bf2f(Q[base + 64 + d]);
    Q[base + d]      = f2bf(q1 * c1 - q2 * s1);
    Q[base + 64 + d] = f2bf(q2 * c2 + q1 * s2);

    float k1 = bf2f(Kk[base + d]), k2 = bf2f(Kk[base + 64 + d]);
    Kk[base + d]      = f2bf(k1 * c1 - k2 * s1);
    Kk[base + 64 + d] = f2bf(k2 * c2 + k1 * s2);
}

// ---------------------------------------------------------------------------
// Flash attention, causal + alibi.  Grid (S/128, H, B), block 256 (8 waves).
// Each wave: 16 q-rows, full D=128 f32 accumulator.  K-block = 32 keys.
// K/V tiles staged via async global->LDS DMA.
// ---------------------------------------------------------------------------
__global__ __launch_bounds__(256) void flash_attn(
    const __bf16* __restrict__ Qb, const __bf16* __restrict__ Kb,
    const __bf16* __restrict__ Vt, const float* __restrict__ alibi,
    __bf16* __restrict__ Ob) {
    __shared__ __bf16 ktile[32 * 128];   // 8 KB  (keys x d, row-major)
    __shared__ __bf16 vtile[128 * 32];   // 8 KB  (d x keys, row-major)
    __shared__ __bf16 pbuf[8][16 * 32];  // 8 KB  per-wave P re-layout bounce

    const int tid  = threadIdx.x;
    const int lane = tid & 31;
    const int wv   = tid >> 5;
    const int b    = blockIdx.z, h = blockIdx.y;
    const int q0   = blockIdx.x * 128 + wv * 16;
    const float nslope = alibi[(size_t)h * Sc + 1];   // == -slope_h

    const __bf16* Qhead = Qb + (size_t)b * Sc * HIDc + (size_t)h * Dc;
    const __bf16* Khead = Kb + (size_t)b * Sc * HIDc + (size_t)h * Dc;
    const __bf16* Vhead = Vt + ((size_t)(b * Hc + h)) * Dc * Sc;

    v16bf qf[4];
#pragma unroll
    for (int dk = 0; dk < 4; ++dk)
        qf[dk] = frag_ld(Qhead, q0, dk * 32, HIDc, lane);

    v8f o[8] = {};
    float mstate[8], lstate[8];
#pragma unroll
    for (int r = 0; r < 8; ++r) { mstate[r] = -1e30f; lstate[r] = 0.f; }

    const int nkb   = (blockIdx.x + 1) * 4;   // causal span in 32-key blocks
    const int qmax  = q0 + 15;
    const int qrb   = (lane >> 4) << 3;       // row offset of this half-wave
    const int kcoll = lane & 15;

    // async-transfer coordinates (32 B per thread per tile)
    const int ki = tid >> 3, kc = (tid & 7) * 16;
    const int vd = tid >> 1, vj = (tid & 1) * 16;
    const unsigned ka = lds_off(&ktile[ki * 128 + kc]);
    const unsigned va = lds_off(&vtile[vd * 32 + vj]);

    for (int kb = 0; kb < nkb; ++kb) {
        const __bf16* kg = &Khead[(size_t)(kb * 32 + ki) * HIDc + kc];
        const __bf16* vg = &Vhead[(size_t)vd * Sc + kb * 32 + vj];
        async_g2l_b128(ka,      kg);
        async_g2l_b128(ka + 16, kg + 8);
        async_g2l_b128(va,      vg);
        async_g2l_b128(va + 16, vg + 8);
        wait_async0();
        __syncthreads();

        if (kb * 32 <= qmax) {           // wave-uniform: EXEC stays all-1s
            v8f c0 = {}, c1 = {};
#pragma unroll
            for (int dk = 0; dk < 4; ++dk) {
                v16bf bk0 = frag_ld(ktile, 0,  dk * 32, 128, lane);
                v16bf bk1 = frag_ld(ktile, 16, dk * 32, 128, lane);
                c0 = wmma_bf16(qf[dk], bk0, c0);
                c1 = wmma_bf16(qf[dk], bk1, c1);
            }
            int kg2 = kb * 32 + kcoll;
            float alpha[8];
#pragma unroll
            for (int r = 0; r < 8; ++r) {
                int qi = q0 + qrb + r;
                float a0 = c0[r] * QK_SCALE + nslope * (float)kg2
                         + ((kg2 > qi) ? -1e9f : 0.f);
                float a1 = c1[r] * QK_SCALE + nslope * (float)(kg2 + 16)
                         + ((kg2 + 16 > qi) ? -1e9f : 0.f);
                float mx = fmaxf(a0, a1);
                mx = fmaxf(mx, __shfl_xor(mx, 8, 16));
                mx = fmaxf(mx, __shfl_xor(mx, 4, 16));
                mx = fmaxf(mx, __shfl_xor(mx, 2, 16));
                mx = fmaxf(mx, __shfl_xor(mx, 1, 16));
                float mnew = fmaxf(mstate[r], mx);
                float al   = __expf(mstate[r] - mnew);
                float p0   = __expf(a0 - mnew);
                float p1   = __expf(a1 - mnew);
                float sum  = p0 + p1;
                sum += __shfl_xor(sum, 8, 16);
                sum += __shfl_xor(sum, 4, 16);
                sum += __shfl_xor(sum, 2, 16);
                sum += __shfl_xor(sum, 1, 16);
                lstate[r] = lstate[r] * al + sum;
                mstate[r] = mnew;
                alpha[r]  = al;
                int prow = qrb + r;
                pbuf[wv][prow * 32 + kcoll]      = f2bf(p0);
                pbuf[wv][prow * 32 + kcoll + 16] = f2bf(p1);
            }
#pragma unroll
            for (int dt = 0; dt < 8; ++dt)
#pragma unroll
                for (int r = 0; r < 8; ++r) o[dt][r] *= alpha[r];

            v16bf pf = frag_ld(pbuf[wv], 0, 0, 32, lane);
#pragma unroll
            for (int dt = 0; dt < 8; ++dt) {
                v16bf vf = frag_ld(vtile, dt * 16, 0, 32, lane);
                o[dt] = wmma_bf16(pf, vf, o[dt]);
            }
        }
        __syncthreads();
    }

    float inv[8];
#pragma unroll
    for (int r = 0; r < 8; ++r) inv[r] = 1.0f / lstate[r];
#pragma unroll
    for (int dt = 0; dt < 8; ++dt) {
#pragma unroll
        for (int r = 0; r < 8; ++r) {
            int srow = q0 + qrb + r;
            int col  = dt * 16 + kcoll;
            Ob[((size_t)(b * Sc + srow)) * HIDc + h * Dc + col] =
                f2bf(o[dt][r] * inv[r]);
        }
    }
}

// ---------------------------------------------------------------------------
// Host orchestration (all launches on `stream`; graph-capture safe).
// ---------------------------------------------------------------------------
extern "C" void kernel_launch(void* const* d_in, const int* in_sizes, int n_in,
                              void* d_out, int out_size, void* d_ws, size_t ws_size,
                              hipStream_t stream) {
    const float* x     = (const float*)d_in[0];
    const float* Wq    = (const float*)d_in[1];
    const float* Wk    = (const float*)d_in[2];
    const float* Wv    = (const float*)d_in[3];
    const float* Wo    = (const float*)d_in[4];
    /* d_in[5] attention_mask: causal, computed analytically */
    const float* alibi = (const float*)d_in[6];
    const float* fc    = (const float*)d_in[7];
    const float* fs    = (const float*)d_in[8];

    const int M  = Bc * Sc;                 // 4096
    const size_t NX = (size_t)M * HIDc;     // 8,388,608
    const size_t NW = (size_t)HIDc * HIDc;  // 4,194,304

    char* ws = (char*)d_ws;
    size_t off = 0;
    auto carve = [&](size_t bytes) -> char* {
        char* p = ws + off;
        off += (bytes + 255) & ~(size_t)255;
        return p;
    };
    __bf16* xb  = (__bf16*)carve(NX * 2);     // also reused as Ob (attn output)
    __bf16* Wqp = (__bf16*)carve(NW * 2);
    __bf16* Wkp = (__bf16*)carve(NW * 2);
    __bf16* Wvp = (__bf16*)carve(NW * 2);
    __bf16* Wop = (__bf16*)carve(NW * 2);
    __bf16* Qb  = (__bf16*)carve(NX * 2);
    __bf16* Kb  = (__bf16*)carve(NX * 2);
    __bf16* Vt  = (__bf16*)carve(NX * 2);
    (void)ws_size; (void)in_sizes; (void)n_in; (void)out_size;

    cast_to_bf16<<<(unsigned)(NX / 4 / 256), 256, 0, stream>>>(x, xb, (int)NX);

    const unsigned packBlocks = (unsigned)(((HIDc / 32) * (HIDc / 16) * 32) / 256); // 1024
    pack_w<<<packBlocks, 256, 0, stream>>>(Wq, Wqp, HIDc, HIDc);
    pack_w<<<packBlocks, 256, 0, stream>>>(Wk, Wkp, HIDc, HIDc);
    pack_w<<<packBlocks, 256, 0, stream>>>(Wv, Wvp, HIDc, HIDc);
    pack_w<<<packBlocks, 256, 0, stream>>>(Wo, Wop, HIDc, HIDc);

    dim3 gg(HIDc / 256, M / 64);   // (8, 64)
    gemm_bf16_wmma<<<gg, 256, 0, stream>>>(xb, Wqp, nullptr, Qb, M, HIDc, HIDc, 0);
    gemm_bf16_wmma<<<gg, 256, 0, stream>>>(xb, Wkp, nullptr, Kb, M, HIDc, HIDc, 0);
    gemm_bf16_wmma<<<gg, 256, 0, stream>>>(xb, Wvp, nullptr, Vt, M, HIDc, HIDc, 1);

    rope_qk<<<(unsigned)((size_t)Bc * Sc * Hc * 64 / 256), 256, 0, stream>>>(Qb, Kb, fc, fs);

    __bf16* Ob = xb;   // x no longer needed; alias to save workspace
    flash_attn<<<dim3(Sc / 128, Hc, Bc), 256, 0, stream>>>(Qb, Kb, Vt, alibi, Ob);

    gemm_bf16_wmma<<<gg, 256, 0, stream>>>(Ob, Wop, (float*)d_out, nullptr, M, HIDc, HIDc, 2);
}
